// LSTM_AD_30657476558920
// MI455X (gfx1250) — matrix operand order repaired and checked by reading, not compiled, not run on hardware
//
#include <hip/hip_runtime.h>
#include <hip/hip_bf16.h>

typedef __attribute__((ext_vector_type(16))) _Float16 v16h;
typedef __attribute__((ext_vector_type(8)))  float    v8f;
typedef __attribute__((ext_vector_type(4)))  float    v4f;
typedef __attribute__((ext_vector_type(2)))  float    v2f;

// ---------------------------------------------------------------------------
// Fast activations: v_exp_f32 + v_rcp_f32 on the transcendental pipe.
// ---------------------------------------------------------------------------
__device__ __forceinline__ float fsig(float x) {
    float e = __expf(-x);
    return __builtin_amdgcn_rcpf(1.0f + e);
}
__device__ __forceinline__ float ftanh(float x) {
    // tanh(x) = 2/(1+exp(-2x)) - 1
    float e = __expf(-2.0f * x);
    float r = __builtin_amdgcn_rcpf(1.0f + e);
    return fmaf(2.0f, r, -1.0f);
}

__device__ __forceinline__ void cvt4(v16h& v, int b, float4 f) {
    v[b + 0] = (_Float16)f.x; v[b + 1] = (_Float16)f.y;
    v[b + 2] = (_Float16)f.z; v[b + 3] = (_Float16)f.w;
}

// ---------------------------------------------------------------------------
// Kernel 1: gx2[T,512] = x[T,128] @ W_ih2[512,128]^T + (b_ih2 + b_hh2)
// One block = 16 rows of T, 8 waves; wave w computes N-tiles {w, w+8, w+16, w+24}.
// A fragment (16x32 f16) loaded once per K-chunk, reused for all 4 N-tiles.
// ---------------------------------------------------------------------------
__global__ __launch_bounds__(256) void gx2_gemm_wmma(
    const float* __restrict__ x, const float* __restrict__ Wih2,
    const float* __restrict__ bih2, const float* __restrict__ bhh2,
    float* __restrict__ gx2)
{
    const int lane   = threadIdx.x & 31;
    const int wave   = threadIdx.x >> 5;
    const int lane16 = lane & 15;
    const int hi     = lane >> 4;           // 0: lanes 0-15, 1: lanes 16-31
    const int m16    = blockIdx.x * 16;     // T-tile base row
    const int mrow   = m16 + lane16;        // A-matrix row supplied by this lane

    // 16-bit A 16x32 layout: lanes 0-15 hold K {0..7, 16..23},
    // lanes 16-31 hold K {8..15, 24..31}.
    v16h a[4];
    const int khalfA = hi * 8;
#pragma unroll
    for (int kk = 0; kk < 4; ++kk) {
        const float* xr = x + (size_t)mrow * 128 + kk * 32 + khalfA;
        cvt4(a[kk], 0,  *(const float4*)(xr + 0));
        cvt4(a[kk], 4,  *(const float4*)(xr + 4));
        cvt4(a[kk], 8,  *(const float4*)(xr + 16));
        cvt4(a[kk], 12, *(const float4*)(xr + 20));
    }

    const int kbaseB = hi * 8;   // B 32x16: lanes 0-15 K {0..7, 16..23}, lanes 16-31 K {8..15, 24..31}
#pragma unroll
    for (int nt = 0; nt < 4; ++nt) {
        const int n16  = (wave + nt * 8) * 16;
        const int ncol = n16 + lane16;      // output gate column == W_ih2 row
        const float bias = bih2[ncol] + bhh2[ncol];

        v8f acc;
#pragma unroll
        for (int r = 0; r < 8; ++r) acc[r] = bias;

#pragma unroll
        for (int kk = 0; kk < 4; ++kk) {
            const float* wr = Wih2 + (size_t)ncol * 128 + kk * 32 + kbaseB;
            v16h b;
            cvt4(b, 0,  *(const float4*)(wr + 0));
            cvt4(b, 4,  *(const float4*)(wr + 4));
            cvt4(b, 8,  *(const float4*)(wr + 16));
            cvt4(b, 12, *(const float4*)(wr + 20));
            acc = __builtin_amdgcn_wmma_f32_16x16x32_f16(
                      false, a[kk], false, b, (short)0, acc, false, false);
        }

        // C layout: VGPR r -> M = r (lanes 0-15) / r+8 (lanes 16-31); N = lane%16
#pragma unroll
        for (int r = 0; r < 8; ++r) {
            const int m = m16 + r + hi * 8;
            gx2[(size_t)m * 512 + ncol] = acc[r];
        }
    }
}

// ---------------------------------------------------------------------------
// Kernel 2: persistent single-workgroup sequential scan over T steps.
// 512 threads (16 waves = 4 waves/SIMD): thread tid owns gate row tid of W_hh2
// (torch order i,f,g,o x128) with its 128 weights resident in VGPRs.
// h2 lives in LDS (uniform-address b128 broadcast reads). Per step: gate
// matvec -> cell update; h_t is streamed to hbuf (= d_out) for the later
// parallel FC GEMM, keeping the FC off the sequential critical path.
// ---------------------------------------------------------------------------
__global__ __launch_bounds__(512) void lstm_scan(
    const float* __restrict__ gx2, const float* __restrict__ Whh2,
    const float* __restrict__ h2_0, const float* __restrict__ c2_0,
    float* __restrict__ hbuf, int T)
{
    __shared__ __align__(16) float h2s[128];   // current hidden state h_{t-1}
    __shared__ float gbuf[512];                // gate pre-activations

    const int tid = threadIdx.x;

    // 128 weights -> 64 x v2f = 128 VGPRs (4 waves/SIMD: no spill pressure).
    v2f w[64];
    const v2f* wp = (const v2f*)(Whh2 + (size_t)tid * 128);
#pragma unroll
    for (int k = 0; k < 64; ++k) w[k] = wp[k];

    float c2 = 0.0f;
    if (tid < 128) { h2s[tid] = h2_0[tid]; c2 = c2_0[tid]; }
    __syncthreads();

    for (int t = 0; t < T; ++t) {
        // Independent gx2 stream load first; prefetch next step's row.
        float g = gx2[(size_t)t * 512 + tid];
        if (t + 1 < T)
            __builtin_prefetch(&gx2[(size_t)(t + 1) * 512 + tid], 0, 0);

        // dot(w_row, h_{t-1}): 32 x ds_load_b128 broadcasts + packed f32 FMAs
        const v4f* h4 = (const v4f*)h2s;
        v2f accA = {0.0f, 0.0f}, accB = {0.0f, 0.0f};
#pragma unroll
        for (int q = 0; q < 32; ++q) {
            v4f hv = h4[q];
            accA = __builtin_elementwise_fma(w[2 * q],     hv.xy, accA);
            accB = __builtin_elementwise_fma(w[2 * q + 1], hv.zw, accB);
        }
        gbuf[tid] = g + (accA.x + accA.y) + (accB.x + accB.y);
        __syncthreads();                       // gbuf ready; h2s reads done

        if (tid < 128) {
            const float gi = fsig (gbuf[tid]);
            const float gf = fsig (gbuf[128 + tid]);
            const float gg = ftanh(gbuf[256 + tid]);
            const float go = fsig (gbuf[384 + tid]);
            c2 = fmaf(gf, c2, gi * gg);
            const float hn = go * ftanh(c2);   // h_t
            h2s[tid] = hn;
            hbuf[(size_t)t * 128 + tid] = hn;  // fire-and-forget store for FC GEMM
        }
        __syncthreads();                       // h2s == h_t
    }
}

// ---------------------------------------------------------------------------
// Kernel 3: out[T,128] = 2*sigmoid(H[T,128] @ W_fc^T + b_fc), IN PLACE on io.
// One wave owns one 16-row stripe: it loads the whole 16x128 input block into
// A fragments (registers) before writing any column of that block, so the
// in-place update has no RAW hazard. 32 v_wmma per wave.
// ---------------------------------------------------------------------------
__global__ __launch_bounds__(256) void fc_gemm_wmma(
    float* io, const float* __restrict__ Wfc, const float* __restrict__ bfc)
{
    const int lane   = threadIdx.x & 31;
    const int wave   = threadIdx.x >> 5;
    const int lane16 = lane & 15;
    const int hi     = lane >> 4;
    const int m16    = (blockIdx.x * 8 + wave) * 16;   // this wave's row stripe
    const int mrow   = m16 + lane16;

    // Load the full 16x128 block of H into A fragments (all reads before writes).
    v16h a[4];
    const int khalfA = hi * 8;
#pragma unroll
    for (int kk = 0; kk < 4; ++kk) {
        const float* xr = io + (size_t)mrow * 128 + kk * 32 + khalfA;
        cvt4(a[kk], 0,  *(const float4*)(xr + 0));
        cvt4(a[kk], 4,  *(const float4*)(xr + 4));
        cvt4(a[kk], 8,  *(const float4*)(xr + 16));
        cvt4(a[kk], 12, *(const float4*)(xr + 20));
    }

    const int kbaseB = hi * 8;
#pragma unroll
    for (int nt = 0; nt < 8; ++nt) {
        const int ncol = nt * 16 + lane16;      // output column == W_fc row
        const float bias = bfc[ncol];

        v8f acc;
#pragma unroll
        for (int r = 0; r < 8; ++r) acc[r] = bias;

#pragma unroll
        for (int kk = 0; kk < 4; ++kk) {
            const float* wr = Wfc + (size_t)ncol * 128 + kk * 32 + kbaseB;
            v16h b;
            cvt4(b, 0,  *(const float4*)(wr + 0));
            cvt4(b, 4,  *(const float4*)(wr + 4));
            cvt4(b, 8,  *(const float4*)(wr + 16));
            cvt4(b, 12, *(const float4*)(wr + 20));
            acc = __builtin_amdgcn_wmma_f32_16x16x32_f16(
                      false, a[kk], false, b, (short)0, acc, false, false);
        }

#pragma unroll
        for (int r = 0; r < 8; ++r) {
            const int m = m16 + r + hi * 8;
            io[(size_t)m * 128 + ncol] = 2.0f * fsig(acc[r]);
        }
    }
}

// ---------------------------------------------------------------------------
// Launch: GEMM fills d_ws with gx2[T,512] (128 MiB -> hot in 192 MB L2),
// the persistent scan consumes it and streams h_t into d_out, then the FC
// WMMA GEMM transforms d_out in place. Same stream -> ordered.
// ---------------------------------------------------------------------------
extern "C" void kernel_launch(void* const* d_in, const int* in_sizes, int n_in,
                              void* d_out, int out_size, void* d_ws, size_t ws_size,
                              hipStream_t stream) {
    const float* x    = (const float*)d_in[0];
    const float* h2_0 = (const float*)d_in[3];
    const float* c2_0 = (const float*)d_in[4];
    // d_in[5..8] (layer-1 weights/biases) are dead w.r.t. the output: skipped.
    const float* Wih2 = (const float*)d_in[9];
    const float* Whh2 = (const float*)d_in[10];
    const float* bih2 = (const float*)d_in[11];
    const float* bhh2 = (const float*)d_in[12];
    const float* Wfc  = (const float*)d_in[13];
    const float* bfc  = (const float*)d_in[14];

    float* outp = (float*)d_out;         // holds H during the scan, then Y
    float* gx2  = (float*)d_ws;          // needs T*512*4 = 128 MiB scratch

    const int T = in_sizes[0] / 128;     // 65536

    gx2_gemm_wmma<<<dim3(T / 16), dim3(256), 0, stream>>>(x, Wih2, bih2, bhh2, gx2);
    lstm_scan<<<dim3(1), dim3(512), 0, stream>>>(gx2, Whh2, h2_0, c2_0, outp, T);
    fc_gemm_wmma<<<dim3(T / 128), dim3(256), 0, stream>>>(outp, Wfc, bfc);
}
